// Generator_27315992003205
// MI455X (gfx1250) — matrix operand order repaired and checked by reading, not compiled, not run on hardware
//
#include <hip/hip_runtime.h>

// ---- types for WMMA ----
typedef _Float16 v16h __attribute__((ext_vector_type(16)));
typedef _Float16 v8h  __attribute__((ext_vector_type(8)));
typedef float    v8f  __attribute__((ext_vector_type(8)));

#define NZd   128
#define NGEN  8
#define BATCH 2048
#define H1D   256
#define H2D   1568   // 32*7*7
#define SLOPE 0.2f

__device__ __forceinline__ float lrelu(float x) { return x > 0.f ? x : SLOPE * x; }

// ---------------- workspace layout (bytes) ----------------
// counts : 0        (32 B)
// lists  : 256      (8*2048*4 = 65536)
// zf16   : 66048    (2048*128*2 = 524288)
// w1t    : 590336   (8*256*128*2 = 524288)   layout [g][n][k] f16
// w2t    : 1114624  (8*1568*256*2 = 6422528) layout [g][n][k] f16
// h2     : 7537152  (2048*1568*2 = 6422528)  f16
#define OFF_COUNTS 0
#define OFF_LISTS  256
#define OFF_ZF16   66048
#define OFF_W1T    590336
#define OFF_W2T    1114624
#define OFF_H2     7537152

// ---------------- kernel 0: zero bin counters ----------------
__global__ void zero_counts_kernel(int* counts) {
    if (threadIdx.x < NGEN) counts[threadIdx.x] = 0;
}

// ---------------- kernel 1: bin samples by generator ----------------
__global__ void bin_kernel(const int* __restrict__ g_idx, int* counts, int* lists) {
    int s = blockIdx.x * blockDim.x + threadIdx.x;
    if (s >= BATCH) return;
    int g = g_idx[s];
    int pos = atomicAdd(&counts[g], 1);
    lists[g * BATCH + pos] = s;
}

// ---------------- kernel 2: f32 -> f16 conversions (+ transpose W to [g][n][k]) ----
__global__ void prep_kernel(const float* __restrict__ z,
                            const float* __restrict__ W1,
                            const float* __restrict__ W2,
                            _Float16* __restrict__ zf16,
                            _Float16* __restrict__ w1t,
                            _Float16* __restrict__ w2t) {
    int i = blockIdx.x * blockDim.x + threadIdx.x;
    const int NZT = BATCH * NZd;            // 262144
    const int N1  = NGEN * H1D * NZd;       // 262144
    const int N2  = NGEN * H2D * H1D;       // 3211264
    if (i < NZT) { zf16[i] = (_Float16)z[i]; return; }
    i -= NZT;
    if (i < N1) {   // w1t[g][n][k] = W1[g][k][n]
        int g = i / (H1D * NZd); int r = i % (H1D * NZd);
        int n = r / NZd; int k = r % NZd;
        w1t[i] = (_Float16)W1[((size_t)g * NZd + k) * H1D + n];
        return;
    }
    i -= N1;
    if (i < N2) {   // w2t[g][n][k] = W2[g][k][n]
        int g = i / (H2D * H1D); int r = i % (H2D * H1D);
        int n = r / H1D; int k = r % H1D;
        w2t[i] = (_Float16)W2[((size_t)g * H1D + k) * H2D + n];
        return;
    }
}

// ---------------- kernel 3: FC1 + FC2 via WMMA (one 16-sample tile / block) ----
__global__ __launch_bounds__(256)
void fc_kernel(const _Float16* __restrict__ zf16,
               const _Float16* __restrict__ w1t,
               const _Float16* __restrict__ w2t,
               const float* __restrict__ b1,
               const float* __restrict__ b2,
               const int* __restrict__ counts,
               const int* __restrict__ lists,
               _Float16* __restrict__ h2) {
    int g = blockIdx.y;
    int t = blockIdx.x;
    int cnt = counts[g];
    if (t * 16 >= cnt) return;                  // uniform per block
    int rem = cnt - t * 16;
    int v = rem < 16 ? rem : 16;

    __shared__ int S[16];
    __shared__ _Float16 h1[16 * H1D];           // 8 KB

    int tid = threadIdx.x;
    if (tid < 16) {
        int m = tid;
        S[m] = lists[g * BATCH + t * 16 + (m < v ? m : 0)];
    }
    __syncthreads();

    int lane = tid & 31;
    int wave = tid >> 5;
    int mloc = lane & 15;       // A row / B,C column within tile
    int hi   = lane >> 4;       // lane-group select
    int kbA  = hi * 8;          // A fragment K offset (16-bit A layout)
    int kbB  = hi * 8;          // B fragment K offset (16-bit B layout)
    int rowz = S[mloc] * NZd;

    // ---- FC1: [16 x 128] @ [128 x 256] -> h1 (f16, LDS), 16 N-tiles / 8 waves
    for (int nt = wave; nt < H1D / 16; nt += 8) {
        int n = nt * 16 + mloc;
        v8f acc = {0.f, 0.f, 0.f, 0.f, 0.f, 0.f, 0.f, 0.f};
        const _Float16* ab = zf16 + rowz + kbA;
        const _Float16* bb = w1t + ((size_t)g * H1D + n) * NZd + kbB;
        #pragma unroll
        for (int kk = 0; kk < NZd; kk += 32) {
            v8h alo = *(const v8h*)(ab + kk);
            v8h ahi = *(const v8h*)(ab + kk + 16);
            v16h a = __builtin_shufflevector(alo, ahi,
                        0,1,2,3,4,5,6,7,8,9,10,11,12,13,14,15);
            v8h blo = *(const v8h*)(bb + kk);
            v8h bhi = *(const v8h*)(bb + kk + 16);
            v16h b = __builtin_shufflevector(blo, bhi,
                        0,1,2,3,4,5,6,7,8,9,10,11,12,13,14,15);
            acc = __builtin_amdgcn_wmma_f32_16x16x32_f16(
                      false, a, false, b, (short)0, acc, false, false);
        }
        float bias = b1[g * H1D + n];
        #pragma unroll
        for (int r = 0; r < 8; ++r) {
            int m = r + hi * 8;
            h1[m * H1D + n] = (_Float16)lrelu(acc[r] + bias);
        }
    }
    __syncthreads();

    // ---- FC2: [16 x 256] @ [256 x 1568] -> h2 (f16, global ws), 98 N-tiles
    for (int nt = wave; nt < H2D / 16; nt += 8) {
        int n = nt * 16 + mloc;
        v8f acc = {0.f, 0.f, 0.f, 0.f, 0.f, 0.f, 0.f, 0.f};
        const _Float16* bb = w2t + ((size_t)g * H2D + n) * H1D + kbB;
        #pragma unroll
        for (int kk = 0; kk < H1D; kk += 32) {
            v8h alo = *(const v8h*)(&h1[mloc * H1D + kk + kbA]);
            v8h ahi = *(const v8h*)(&h1[mloc * H1D + kk + kbA + 16]);
            v16h a = __builtin_shufflevector(alo, ahi,
                        0,1,2,3,4,5,6,7,8,9,10,11,12,13,14,15);
            v8h blo = *(const v8h*)(bb + kk);
            v8h bhi = *(const v8h*)(bb + kk + 16);
            v16h b = __builtin_shufflevector(blo, bhi,
                        0,1,2,3,4,5,6,7,8,9,10,11,12,13,14,15);
            acc = __builtin_amdgcn_wmma_f32_16x16x32_f16(
                      false, a, false, b, (short)0, acc, false, false);
        }
        float bias = b2[g * H2D + n];
        #pragma unroll
        for (int r = 0; r < 8; ++r) {
            int m = r + hi * 8;
            if (m < v)
                h2[(size_t)S[m] * H2D + n] = (_Float16)lrelu(acc[r] + bias);
        }
    }
}

// ---------------- kernel 4: fused conv-transpose tail, one sample / block ----
__global__ __launch_bounds__(128)
void conv_kernel(const _Float16* __restrict__ h2,
                 const int* __restrict__ g_idx,
                 const float* __restrict__ cw1, const float* __restrict__ cb1,
                 const float* __restrict__ cw2, const float* __restrict__ cb2,
                 const float* __restrict__ cw3, const float* __restrict__ cb3,
                 float* __restrict__ out) {
    int s = blockIdx.x;
    int g = g_idx[s];
    int tid = threadIdx.x;

    __shared__ float x0[32 * 7 * 7];       // 6.1 KB
    __shared__ float a1[16 * 14 * 14];     // 12.3 KB
    __shared__ float a2[8 * 28 * 28];      // 24.5 KB

    for (int i = tid; i < H2D; i += 128)
        x0[i] = (float)h2[(size_t)s * H2D + i];
    __syncthreads();

    // convT1: 32ch 7x7 -> 16ch 14x14, k=4, s=2, effective torch-pad=1
    for (int o = tid; o < 16 * 196; o += 128) {
        int co = o / 196, r = o % 196, oy = r / 14, ox = r % 14;
        float acc = cb1[g * 16 + co];
        for (int ky = 0; ky < 4; ++ky) {
            int ty = oy + 1 - ky;
            if (ty < 0 || (ty & 1)) continue;
            int iy = ty >> 1; if (iy >= 7) continue;
            for (int kx = 0; kx < 4; ++kx) {
                int tx = ox + 1 - kx;
                if (tx < 0 || (tx & 1)) continue;
                int ix = tx >> 1; if (ix >= 7) continue;
                const float* xp = &x0[iy * 7 + ix];
                const float* wq = cw1 + (size_t)g * 8192 + co * 16 + ky * 4 + kx;
                float a = 0.f;
                #pragma unroll 8
                for (int ci = 0; ci < 32; ++ci) a += xp[ci * 49] * wq[ci * 256];
                acc += a;
            }
        }
        a1[o] = lrelu(acc);
    }
    __syncthreads();

    // convT2: 16ch 14x14 -> 8ch 28x28, k=4, s=2, pad=1
    for (int o = tid; o < 8 * 784; o += 128) {
        int co = o / 784, r = o % 784, oy = r / 28, ox = r % 28;
        float acc = cb2[g * 8 + co];
        for (int ky = 0; ky < 4; ++ky) {
            int ty = oy + 1 - ky;
            if (ty < 0 || (ty & 1)) continue;
            int iy = ty >> 1; if (iy >= 14) continue;
            for (int kx = 0; kx < 4; ++kx) {
                int tx = ox + 1 - kx;
                if (tx < 0 || (tx & 1)) continue;
                int ix = tx >> 1; if (ix >= 14) continue;
                const float* xp = &a1[iy * 14 + ix];
                const float* wq = cw2 + (size_t)g * 2048 + co * 16 + ky * 4 + kx;
                float a = 0.f;
                #pragma unroll 8
                for (int ci = 0; ci < 16; ++ci) a += xp[ci * 196] * wq[ci * 128];
                acc += a;
            }
        }
        a2[o] = lrelu(acc);
    }
    __syncthreads();

    // convT3: 8ch 28x28 -> 1ch 28x28, k=3, s=1, pad=1, then tanh
    for (int o = tid; o < 784; o += 128) {
        int oy = o / 28, ox = o % 28;
        float acc = cb3[g];
        for (int ky = 0; ky < 3; ++ky) {
            int iy = oy + 1 - ky;
            if (iy < 0 || iy >= 28) continue;
            for (int kx = 0; kx < 3; ++kx) {
                int ix = ox + 1 - kx;
                if (ix < 0 || ix >= 28) continue;
                const float* wq = cw3 + (size_t)g * 72 + ky * 3 + kx;
                float a = 0.f;
                #pragma unroll
                for (int ci = 0; ci < 8; ++ci)
                    a += a2[ci * 784 + iy * 28 + ix] * wq[ci * 9];
                acc += a;
            }
        }
        out[(size_t)s * 784 + o] = tanhf(acc);
    }
}

// ---------------- launcher ----------------
extern "C" void kernel_launch(void* const* d_in, const int* in_sizes, int n_in,
                              void* d_out, int out_size, void* d_ws, size_t ws_size,
                              hipStream_t stream) {
    const float* z   = (const float*)d_in[0];
    const int*   gix = (const int*)  d_in[1];
    const float* W1  = (const float*)d_in[2];
    const float* b1  = (const float*)d_in[3];
    const float* W2  = (const float*)d_in[4];
    const float* b2  = (const float*)d_in[5];
    const float* cw1 = (const float*)d_in[6];
    const float* cb1 = (const float*)d_in[7];
    const float* cw2 = (const float*)d_in[8];
    const float* cb2 = (const float*)d_in[9];
    const float* cw3 = (const float*)d_in[10];
    const float* cb3 = (const float*)d_in[11];
    float* out = (float*)d_out;

    char* ws = (char*)d_ws;
    int*       counts = (int*)      (ws + OFF_COUNTS);
    int*       lists  = (int*)      (ws + OFF_LISTS);
    _Float16*  zf16   = (_Float16*) (ws + OFF_ZF16);
    _Float16*  w1t    = (_Float16*) (ws + OFF_W1T);
    _Float16*  w2t    = (_Float16*) (ws + OFF_W2T);
    _Float16*  h2     = (_Float16*) (ws + OFF_H2);

    zero_counts_kernel<<<1, 32, 0, stream>>>(counts);
    bin_kernel<<<BATCH / 256, 256, 0, stream>>>(gix, counts, lists);

    int totalPrep = BATCH * NZd + NGEN * H1D * NZd + NGEN * H2D * H1D;
    prep_kernel<<<(totalPrep + 255) / 256, 256, 0, stream>>>(z, W1, W2, zf16, w1t, w2t);

    dim3 fgrid(BATCH / 16, NGEN);   // early-exit on tiles past per-generator count
    fc_kernel<<<fgrid, 256, 0, stream>>>(zf16, w1t, w2t, b1, b2, counts, lists, h2);

    conv_kernel<<<BATCH, 128, 0, stream>>>(h2, gix, cw1, cb1, cw2, cb2, cw3, cb3, out);
}
